// RNNEncoder_74947179315933
// MI455X (gfx1250) — matrix operand order repaired and checked by reading, not compiled, not run on hardware
//
#include <hip/hip_runtime.h>
#include <hip/hip_bf16.h>
#include <stddef.h>

// ---------------------------------------------------------------------------
// Types for WMMA fragments (wave32, v_wmma_f32_16x16x32_f16)
// ---------------------------------------------------------------------------
typedef _Float16 half8  __attribute__((ext_vector_type(8)));
typedef _Float16 half16 __attribute__((ext_vector_type(16)));
typedef float    floatx8 __attribute__((ext_vector_type(8)));

union H16U { half16 v16; half8 v8[2]; };

// ---------------------------------------------------------------------------
// Workspace layout (bytes). Total ~12.1 MB.
// ---------------------------------------------------------------------------
static constexpr size_t OFF_WMI = 0;                       // 128*1024*2  = 262144
static constexpr size_t OFF_WMH = OFF_WMI + 262144;        // 256*1024*2  = 524288
static constexpr size_t OFF_WE  = OFF_WMH + 524288;        // 512*128*2   = 131072
static constexpr size_t OFF_WEI = OFF_WE  + 131072;        // 128*1024*2  = 262144
static constexpr size_t OFF_WEH = OFF_WEI + 262144;        // 256*1024*2  = 524288
static constexpr size_t OFF_HM  = OFF_WEH + 524288;        // 320*63*256*2 = 10321920
static constexpr size_t OFF_A   = OFF_HM  + 10321920;      // 6080*4

// ---------------------------------------------------------------------------
// Fragment loaders.
//
// 16-bit A matrix 16x32 (MxK), ISA 7.12.2:
//   lane L: m = L%16, hi = L/16
//   halfs j=0..7  -> K = 8*hi + j          (contiguous 16B chunk)
//   halfs j=8..15 -> K = 16 + 8*hi + (j-8) (contiguous 16B chunk)
// Source is row-major [rows x strideHalfs] f16 in LDS.
// ---------------------------------------------------------------------------
__device__ __forceinline__ half16 load_a_frag(const _Float16* __restrict__ src,
                                              int strideHalfs, int kbase, int lane) {
    const int m  = lane & 15;
    const int hi = lane >> 4;
    const _Float16* row = src + m * strideHalfs + kbase + 8 * hi;
    H16U u;
    u.v8[0] = *(const half8*)(row);
    u.v8[1] = *(const half8*)(row + 16);
    return u.v16;
}

// B matrix 32x16 (KxN), packed by pack_b_kernel so that each lane's 16 halfs
// are contiguous: packed[((kt*NT+nt)*32 + lane)*16 + j], with
//   n = lane%16, k = kt*32 + 8*(lane/16) + (j%8) + 16*(j/8)
__device__ __forceinline__ half16 load_b_frag(const _Float16* __restrict__ packed,
                                              int tileIdx, int lane) {
    const _Float16* p = packed + ((size_t)tileIdx * 32 + lane) * 16;
    H16U u;
    u.v8[0] = *(const half8*)(p);
    u.v8[1] = *(const half8*)(p + 8);
    return u.v16;
}

__device__ __forceinline__ floatx8 wmma_f16(half16 a, half16 b, floatx8 c) {
    return __builtin_amdgcn_wmma_f32_16x16x32_f16(false, a, false, b,
                                                  (short)0, c, false, false);
}

__device__ __forceinline__ float sigmoidf_(float x) {
    return 1.0f / (1.0f + __expf(-x));
}
__device__ __forceinline__ float tanhf_(float x) {
    x = fminf(fmaxf(x, -15.0f), 15.0f);
    float e = __expf(2.0f * x);
    return (e - 1.0f) / (e + 1.0f);
}

// ---------------------------------------------------------------------------
// Kernel 0: pack a row-major fp32 [K x N] weight into WMMA-B f16 tile order.
// ---------------------------------------------------------------------------
__global__ void pack_b_kernel(const float* __restrict__ W, _Float16* __restrict__ dst,
                              int K, int N) {
    int p = blockIdx.x * blockDim.x + threadIdx.x;
    if (p >= K * N) return;
    int j      = p & 15;
    int lane   = (p >> 4) & 31;
    int tileid = p >> 9;
    int NT = N >> 4;
    int nt = tileid % NT;
    int kt = tileid / NT;
    int k = kt * 32 + 8 * (lane >> 4) + (j & 7) + 16 * (j >> 3);
    int n = nt * 16 + (lane & 15);
    dst[p] = (_Float16)W[(size_t)k * N + n];
}

// ---------------------------------------------------------------------------
// Kernel 1: motion LSTM. 10 blocks x 256 threads (8 waves).
// Block handles 32 (b,node) samples (two M-tiles) for all 63 steps; h in LDS,
// c in regs. Each wave owns 8 gate-column tiles (all 4 gates for a 32-wide
// hidden slice) so the LSTM update is per-lane register math. Each B fragment
// is loaded once and feeds TWO WMMAs (M-tile 0 and 1) -> halves L2 B-traffic.
// k-tile loops deliberately NOT unrolled (no LICM hoist of loop-invariant
// weight loads -> no VGPR blowup / scratch spills).
// ---------------------------------------------------------------------------
__global__ __launch_bounds__(256, 2) void motion_lstm_kernel(
    const float* __restrict__ X,
    const float* __restrict__ Wm, const float* __restrict__ bm,
    const _Float16* __restrict__ WmiP, const float* __restrict__ bmi,
    const _Float16* __restrict__ WmhP, const float* __restrict__ bmh,
    _Float16* __restrict__ hm) {
    __shared__ _Float16 sh_em[32 * 128];
    __shared__ _Float16 sh_h[32 * 256];
    __shared__ float    sh_wm[4 * 128];
    __shared__ float    sh_bm[128];
    __shared__ float    sh_dx[32 * 4];

    const int tid  = threadIdx.x;
    const int lane = tid & 31;
    const int w    = tid >> 5;
    const int hi   = lane >> 4;
    const int lm   = lane & 15;
    const int smp0 = blockIdx.x * 32;

    for (int i = tid; i < 512; i += 256) sh_wm[i] = Wm[i];
    if (tid < 128) sh_bm[tid] = bm[tid];
    for (int i = tid; i < 32 * 256; i += 256) sh_h[i] = (_Float16)0.0f;

    float biasv[8];
    #pragma unroll
    for (int g = 0; g < 4; ++g)
        #pragma unroll
        for (int s = 0; s < 2; ++s) {
            int col = g * 256 + w * 32 + s * 16 + lm;
            biasv[g * 2 + s] = bmi[col] + bmh[col];
        }

    float creg[32];
    #pragma unroll
    for (int i = 0; i < 32; ++i) creg[i] = 0.0f;

    __syncthreads();

    for (int t = 0; t < 63; ++t) {
        // ---- stage 1a: temporal delta dX for this step
        if (tid < 128) {
            int r = tid >> 2, d = tid & 3;
            const float* xp = X + ((size_t)(smp0 + r) * 64 + t) * 4 + d;
            sh_dx[r * 4 + d] = xp[4] - xp[0];
        }
        __syncthreads();
        // ---- stage 1b: em = dX @ Wm + bm  (K=4, VALU)
        {
            int r  = tid >> 3;
            int c0 = (tid & 7) * 16;
            float d0 = sh_dx[r * 4 + 0], d1 = sh_dx[r * 4 + 1];
            float d2 = sh_dx[r * 4 + 2], d3 = sh_dx[r * 4 + 3];
            #pragma unroll
            for (int q = 0; q < 16; ++q) {
                int c = c0 + q;
                float v = sh_bm[c] + d0 * sh_wm[c] + d1 * sh_wm[128 + c] +
                          d2 * sh_wm[256 + c] + d3 * sh_wm[384 + c];
                sh_em[r * 128 + c] = (_Float16)v;
            }
        }
        __syncthreads();
        // ---- stage 2: WMMA gate preactivations.
        // acc[(g*2+s)*2 + u2] : column tile (g,s), M-tile u2.
        floatx8 acc[16];
        #pragma unroll
        for (int ti = 0; ti < 8; ++ti) {
            floatx8 a;
            #pragma unroll
            for (int r2 = 0; r2 < 8; ++r2) a[r2] = biasv[ti];
            acc[ti * 2 + 0] = a;
            acc[ti * 2 + 1] = a;
        }
        #pragma clang loop unroll(disable)
        for (int kt = 0; kt < 4; ++kt) {            // em @ Wmi  (K=128)
            half16 af0 = load_a_frag(sh_em,            128, kt * 32, lane);
            half16 af1 = load_a_frag(sh_em + 16 * 128, 128, kt * 32, lane);
            #pragma unroll
            for (int g = 0; g < 4; ++g)
                #pragma unroll
                for (int s = 0; s < 2; ++s) {
                    int tc = g * 16 + w * 2 + s;
                    half16 bf_ = load_b_frag(WmiP, kt * 64 + tc, lane);
                    int ai = (g * 2 + s) * 2;
                    acc[ai + 0] = wmma_f16(af0, bf_, acc[ai + 0]);
                    acc[ai + 1] = wmma_f16(af1, bf_, acc[ai + 1]);
                }
        }
        #pragma clang loop unroll(disable)
        for (int kt = 0; kt < 8; ++kt) {            // h @ Wmh  (K=256)
            half16 af0 = load_a_frag(sh_h,            256, kt * 32, lane);
            half16 af1 = load_a_frag(sh_h + 16 * 256, 256, kt * 32, lane);
            #pragma unroll
            for (int g = 0; g < 4; ++g)
                #pragma unroll
                for (int s = 0; s < 2; ++s) {
                    int tc = g * 16 + w * 2 + s;
                    half16 bf_ = load_b_frag(WmhP, kt * 64 + tc, lane);
                    int ai = (g * 2 + s) * 2;
                    acc[ai + 0] = wmma_f16(af0, bf_, acc[ai + 0]);
                    acc[ai + 1] = wmma_f16(af1, bf_, acc[ai + 1]);
                }
        }
        __syncthreads();
        // ---- stage 3: per-lane LSTM cell update (i,f,g,o in matching slots)
        #pragma unroll
        for (int u2 = 0; u2 < 2; ++u2)
            #pragma unroll
            for (int s = 0; s < 2; ++s)
                #pragma unroll
                for (int r = 0; r < 8; ++r) {
                    float ig = sigmoidf_(acc[(0 + s) * 2 + u2][r]);
                    float fg = sigmoidf_(acc[(2 + s) * 2 + u2][r]);
                    float gg = tanhf_(acc[(4 + s) * 2 + u2][r]);
                    float og = sigmoidf_(acc[(6 + s) * 2 + u2][r]);
                    int ci = u2 * 16 + s * 8 + r;
                    float c = fg * creg[ci] + ig * gg;
                    creg[ci] = c;
                    float h = og * tanhf_(c);
                    sh_h[(r + 8 * hi + 16 * u2) * 256 + w * 32 + s * 16 + lm] =
                        (_Float16)h;
                }
        __syncthreads();
        // ---- stage 4: export h to hm[smp][t][:]
        {
            int r  = tid >> 3;
            int c0 = (tid & 7) * 32;
            _Float16* dst = hm + ((size_t)(smp0 + r) * 63 + t) * 256 + c0;
            const half8* src = (const half8*)(sh_h + r * 256 + c0);
            #pragma unroll
            for (int q = 0; q < 4; ++q) ((half8*)dst)[q] = src[q];
        }
        __syncthreads();
    }
}

// ---------------------------------------------------------------------------
// Kernel 2: edge LSTM. 190 blocks x 256 threads; block handles 32 edges.
// Per step: gather [32x512] sender|receiver h, ea = cat@We+be (WMMA),
// gates = (bei+beh) + ea@Wei + h@Weh (WMMA), LSTM update. Final Wf dot+sigmoid.
// B fragments shared across the two M-tiles; next-step gather prefetched.
// ---------------------------------------------------------------------------
__global__ __launch_bounds__(256, 2) void edge_lstm_kernel(
    const _Float16* __restrict__ hm,
    const _Float16* __restrict__ WeP,  const float* __restrict__ be,
    const _Float16* __restrict__ WeiP, const float* __restrict__ bei,
    const _Float16* __restrict__ WehP, const float* __restrict__ beh,
    const float* __restrict__ Wf, const float* __restrict__ bf,
    float* __restrict__ Aout) {
    __shared__ _Float16 sh_cat[32 * 512];
    __shared__ _Float16 sh_ea[32 * 128];
    __shared__ _Float16 sh_h[32 * 256];
    __shared__ float    sh_red[256];

    const int tid  = threadIdx.x;
    const int lane = tid & 31;
    const int w    = tid >> 5;
    const int hi   = lane >> 4;
    const int lm   = lane & 15;
    const int es0  = blockIdx.x * 32;

    for (int i = tid; i < 32 * 256; i += 256) sh_h[i] = (_Float16)0.0f;

    const float bias1 = be[w * 16 + lm];
    float biasv[8];
    #pragma unroll
    for (int g = 0; g < 4; ++g)
        #pragma unroll
        for (int s = 0; s < 2; ++s) {
            int col = g * 256 + w * 32 + s * 16 + lm;
            biasv[g * 2 + s] = bei[col] + beh[col];
        }

    float creg[32];
    #pragma unroll
    for (int i = 0; i < 32; ++i) creg[i] = 0.0f;

    // Gather mapping: row = tid>>3 (edge in block), part = tid&7 (64 halfs).
    const int grow = tid >> 3;
    const int es   = es0 + grow;
    const int b    = es / 380;
    const int e    = es - b * 380;
    const int isnd = e / 19;
    const int kk   = e - isnd * 19;
    const int jrc  = kk + (kk >= isnd ? 1 : 0);
    const int part = tid & 7;
    const size_t base_s = (size_t)(b * 20 + isnd) * 63 * 256;
    const size_t base_r = (size_t)(b * 20 + jrc)  * 63 * 256;
    const size_t src_base = (part < 4) ? (base_s + (size_t)part * 64)
                                       : (base_r + (size_t)(part - 4) * 64);
    _Float16* const cat_dst = sh_cat + grow * 512 + part * 64;

    __syncthreads();

    for (int t = 0; t < 63; ++t) {
        // ---- gather [senders | receivers] at step t (64 halfs per thread)
        {
            const half8* src = (const half8*)(hm + src_base + (size_t)t * 256);
            half8* dst = (half8*)cat_dst;
            #pragma unroll
            for (int q = 0; q < 8; ++q) dst[q] = src[q];
        }
        // prefetch next step's rows into cache while GEMMs run
        if (t + 1 < 63)
            __builtin_prefetch(hm + src_base + (size_t)(t + 1) * 256, 0, 0);
        __syncthreads();
        // ---- GEMM1: ea = cat @ We + be  (K=512, wave w owns column tile w)
        {
            floatx8 acc0, acc1v;
            #pragma unroll
            for (int r = 0; r < 8; ++r) { acc0[r] = bias1; acc1v[r] = bias1; }
            #pragma clang loop unroll(disable)
            for (int kt = 0; kt < 16; ++kt) {
                half16 af0 = load_a_frag(sh_cat,            512, kt * 32, lane);
                half16 af1 = load_a_frag(sh_cat + 16 * 512, 512, kt * 32, lane);
                half16 bf_ = load_b_frag(WeP, kt * 8 + w, lane);
                acc0  = wmma_f16(af0, bf_, acc0);
                acc1v = wmma_f16(af1, bf_, acc1v);
            }
            #pragma unroll
            for (int r = 0; r < 8; ++r) {
                sh_ea[(r + 8 * hi) * 128 + w * 16 + lm]      = (_Float16)acc0[r];
                sh_ea[(r + 8 * hi + 16) * 128 + w * 16 + lm] = (_Float16)acc1v[r];
            }
        }
        __syncthreads();
        // ---- GEMM2: gates = bias + ea@Wei + h@Weh
        floatx8 acc[16];
        #pragma unroll
        for (int ti = 0; ti < 8; ++ti) {
            floatx8 a;
            #pragma unroll
            for (int r2 = 0; r2 < 8; ++r2) a[r2] = biasv[ti];
            acc[ti * 2 + 0] = a;
            acc[ti * 2 + 1] = a;
        }
        #pragma clang loop unroll(disable)
        for (int kt = 0; kt < 4; ++kt) {            // ea @ Wei (K=128)
            half16 af0 = load_a_frag(sh_ea,            128, kt * 32, lane);
            half16 af1 = load_a_frag(sh_ea + 16 * 128, 128, kt * 32, lane);
            #pragma unroll
            for (int g = 0; g < 4; ++g)
                #pragma unroll
                for (int s = 0; s < 2; ++s) {
                    int tc = g * 16 + w * 2 + s;
                    half16 bf_ = load_b_frag(WeiP, kt * 64 + tc, lane);
                    int ai = (g * 2 + s) * 2;
                    acc[ai + 0] = wmma_f16(af0, bf_, acc[ai + 0]);
                    acc[ai + 1] = wmma_f16(af1, bf_, acc[ai + 1]);
                }
        }
        #pragma clang loop unroll(disable)
        for (int kt = 0; kt < 8; ++kt) {            // h @ Weh (K=256)
            half16 af0 = load_a_frag(sh_h,            256, kt * 32, lane);
            half16 af1 = load_a_frag(sh_h + 16 * 256, 256, kt * 32, lane);
            #pragma unroll
            for (int g = 0; g < 4; ++g)
                #pragma unroll
                for (int s = 0; s < 2; ++s) {
                    int tc = g * 16 + w * 2 + s;
                    half16 bf_ = load_b_frag(WehP, kt * 64 + tc, lane);
                    int ai = (g * 2 + s) * 2;
                    acc[ai + 0] = wmma_f16(af0, bf_, acc[ai + 0]);
                    acc[ai + 1] = wmma_f16(af1, bf_, acc[ai + 1]);
                }
        }
        __syncthreads();
        // ---- LSTM cell update
        #pragma unroll
        for (int u2 = 0; u2 < 2; ++u2)
            #pragma unroll
            for (int s = 0; s < 2; ++s)
                #pragma unroll
                for (int r = 0; r < 8; ++r) {
                    float ig = sigmoidf_(acc[(0 + s) * 2 + u2][r]);
                    float fg = sigmoidf_(acc[(2 + s) * 2 + u2][r]);
                    float gg = tanhf_(acc[(4 + s) * 2 + u2][r]);
                    float og = sigmoidf_(acc[(6 + s) * 2 + u2][r]);
                    int ci = u2 * 16 + s * 8 + r;
                    float c = fg * creg[ci] + ig * gg;
                    creg[ci] = c;
                    float h = og * tanhf_(c);
                    sh_h[(r + 8 * hi + 16 * u2) * 256 + w * 32 + s * 16 + lm] =
                        (_Float16)h;
                }
        __syncthreads();
    }

    // ---- final: A[es] = sigmoid(h_last . Wf + bf)
    {
        int row = tid >> 3, seg = tid & 7;
        float p = 0.0f;
        #pragma unroll
        for (int q = 0; q < 32; ++q)
            p += (float)sh_h[row * 256 + seg * 32 + q] * Wf[seg * 32 + q];
        sh_red[tid] = p;
        __syncthreads();
        if (tid < 32) {
            float ssum = 0.0f;
            #pragma unroll
            for (int q = 0; q < 8; ++q) ssum += sh_red[tid * 8 + q];
            Aout[es0 + tid] = sigmoidf_(ssum + bf[0]);
        }
    }
}

// ---------------------------------------------------------------------------
// Kernel 3: scatter edge scores to symmetric dense adjacency [16,20,20].
// ---------------------------------------------------------------------------
__global__ void scatter_kernel(const float* __restrict__ A, float* __restrict__ out) {
    int idx = blockIdx.x * blockDim.x + threadIdx.x;
    if (idx >= 16 * 400) return;
    int b = idx / 400;
    int rem = idx - b * 400;
    int i = rem / 20, j = rem - (rem / 20) * 20;
    float v = 0.0f;
    if (i != j) {
        int e1 = i * 19 + j - (j > i ? 1 : 0);
        int e2 = j * 19 + i - (i > j ? 1 : 0);
        v = 0.5f * (A[b * 380 + e1] + A[b * 380 + e2]);
    }
    out[idx] = v;
}

// ---------------------------------------------------------------------------
// Launch
// ---------------------------------------------------------------------------
extern "C" void kernel_launch(void* const* d_in, const int* in_sizes, int n_in,
                              void* d_out, int out_size, void* d_ws, size_t ws_size,
                              hipStream_t stream) {
    (void)in_sizes; (void)n_in; (void)out_size; (void)ws_size;
    const float* X   = (const float*)d_in[0];
    const float* Wm  = (const float*)d_in[3];
    const float* bm  = (const float*)d_in[4];
    const float* Wmi = (const float*)d_in[5];
    const float* bmi = (const float*)d_in[6];
    const float* Wmh = (const float*)d_in[7];
    const float* bmh = (const float*)d_in[8];
    const float* We  = (const float*)d_in[9];
    const float* be  = (const float*)d_in[10];
    const float* Wei = (const float*)d_in[11];
    const float* bei = (const float*)d_in[12];
    const float* Weh = (const float*)d_in[13];
    const float* beh = (const float*)d_in[14];
    const float* Wf  = (const float*)d_in[15];
    const float* bf  = (const float*)d_in[16];

    char* ws = (char*)d_ws;
    _Float16* WmiP = (_Float16*)(ws + OFF_WMI);
    _Float16* WmhP = (_Float16*)(ws + OFF_WMH);
    _Float16* WeP  = (_Float16*)(ws + OFF_WE);
    _Float16* WeiP = (_Float16*)(ws + OFF_WEI);
    _Float16* WehP = (_Float16*)(ws + OFF_WEH);
    _Float16* hmP  = (_Float16*)(ws + OFF_HM);
    float*    Aws  = (float*)(ws + OFF_A);

    pack_b_kernel<<<(128 * 1024) / 256, 256, 0, stream>>>(Wmi, WmiP, 128, 1024);
    pack_b_kernel<<<(256 * 1024) / 256, 256, 0, stream>>>(Wmh, WmhP, 256, 1024);
    pack_b_kernel<<<(512 * 128)  / 256, 256, 0, stream>>>(We,  WeP,  512, 128);
    pack_b_kernel<<<(128 * 1024) / 256, 256, 0, stream>>>(Wei, WeiP, 128, 1024);
    pack_b_kernel<<<(256 * 1024) / 256, 256, 0, stream>>>(Weh, WehP, 256, 1024);

    motion_lstm_kernel<<<10, 256, 0, stream>>>(X, Wm, bm, WmiP, bmi, WmhP, bmh, hmP);
    edge_lstm_kernel<<<190, 256, 0, stream>>>(hmP, WeP, be, WeiP, bei, WehP, beh,
                                              Wf, bf, Aws);
    scatter_kernel<<<25, 256, 0, stream>>>(Aws, (float*)d_out);
}
